// NodeDecoder_62947040690365
// MI455X (gfx1250) — compile-verified, matrix operand order checked by reading, not run on hardware
//
#include <hip/hip_runtime.h>

#define N_C 25000
#define N_F 100000
#define E_C 100000
#define E_F 400000

typedef unsigned int   u32;
typedef unsigned short u16;
typedef __attribute__((ext_vector_type(16))) __bf16 v16bf;
typedef __attribute__((ext_vector_type(8)))  float  v8f;
typedef __attribute__((ext_vector_type(4)))  u32    v4u;

union FragBF {
  v4u   q[2];
  v16bf v;
};

__device__ __forceinline__ u16 f2bf(float x) {
  u32 u = __float_as_uint(x);
  u32 r = u + 0x7FFFu + ((u >> 16) & 1u);   // round-to-nearest-even
  return (u16)(r >> 16);
}

__device__ __forceinline__ float selu_f(float x) {
  const float sc = 1.0507009873554805f;
  const float al = 1.6732632423543772f;
  return x > 0.f ? sc * x : sc * al * (__expf(x) - 1.f);
}

// ---------------------------------------------------------------------------
// Fused gather + GEMM(bf16 WMMA, f32 acc) + bias + rank1 + SELU + residual.
// Y(n x 128) = act( concat(X0[idx0],X1[idx1],X2[idx2]) @ W + bias + es*ew ) + res
// Wt is the weight pre-transposed/bf16: Wt[col*K + k], col in [0,128).
//
// Block = 256 threads = 8 waves covering a 64-row x 128-col output strip.
// Each wave computes a 32x32 tile (2x2 of 16x16 WMMA tiles, 4 accumulators),
// so each k-step issues 4 LDS + 4 global b128 loads feeding 4 independent
// v_wmma_f32_16x16x32_bf16 ops. K/32 is always even -> k-loop steps by 64
// with two fragment sets in flight for latency hiding.
// ---------------------------------------------------------------------------
__global__ __launch_bounds__(256) void gemm128_wmma(
    const float* __restrict__ X0, const int* __restrict__ idx0,
    const float* __restrict__ X1, const int* __restrict__ idx1,
    const float* __restrict__ X2, const int* __restrict__ idx2,
    int n, int K,
    const u16* __restrict__ Wt,
    const float* __restrict__ bias,
    const float* __restrict__ extra_s, const float* __restrict__ extra_w,
    const float* __restrict__ res,
    float* __restrict__ Y, int act)
{
  __shared__ __align__(16) u16 lds_a[64 * 384];   // 48 KB

  const int tid  = threadIdx.x;
  const int lane = tid & 31;
  const int wave = tid >> 5;
  const int row0 = blockIdx.x * 64;

  // ---- stage A tile (64 x K) as bf16 into LDS, seg-major (seg is uniform)
  const int nseg = K >> 7;
  for (int seg = 0; seg < nseg; ++seg) {
    const float* X  = (seg == 0) ? X0 : (seg == 1 ? X1 : X2);
    const int* idx  = (seg == 0) ? idx0 : (seg == 1 ? idx1 : idx2);
    for (int t = tid; t < 64 * 128; t += 256) {
      int r    = t >> 7;          // kk = t & 127 is constant per thread
      int kk   = t & 127;
      int grow = row0 + r;
      float val = 0.f;
      if (grow < n) {
        int srow = idx ? idx[grow] : grow;
        val = X[(long)srow * 128 + kk];
      }
      lds_a[r * K + (seg << 7) + kk] = f2bf(val);
    }
  }
  __syncthreads();

  const int nrow = lane & 15;     // row within 16-tile (A) / col within 16-tile (B)
  const int hi   = lane >> 4;     // lanes 16-31 take the +8 k-offset halves
  const int rh   = wave >> 2;     // row half: 0 or 1 (32 rows each)
  const int cs   = wave & 3;      // col strip: 0..3 (32 cols each)
  const int r0   = rh * 32;       // local row base of this wave's tile
  const int col0 = cs * 32 + nrow;

  const u16* arow0 = &lds_a[(r0 + nrow) * K];        // A rows for tile row 0
  const u16* arow1 = &lds_a[(r0 + 16 + nrow) * K];   // A rows for tile row 1
  const u16* wcol0 = Wt + (long)col0 * K;
  const u16* wcol1 = Wt + (long)(col0 + 16) * K;

  v8f c00 = {}, c01 = {}, c10 = {}, c11 = {};

  for (int k0 = 0; k0 < K; k0 += 64) {
    const int bA = k0 + hi * 8;
    const int bB = k0 + 32 + hi * 8;
    FragBF a0A, a1A, b0A, b1A, a0B, a1B, b0B, b1B;
    // set A (k0)
    a0A.q[0] = *reinterpret_cast<const v4u*>(arow0 + bA);
    a0A.q[1] = *reinterpret_cast<const v4u*>(arow0 + bA + 16);
    a1A.q[0] = *reinterpret_cast<const v4u*>(arow1 + bA);
    a1A.q[1] = *reinterpret_cast<const v4u*>(arow1 + bA + 16);
    b0A.q[0] = *reinterpret_cast<const v4u*>(wcol0 + bA);
    b0A.q[1] = *reinterpret_cast<const v4u*>(wcol0 + bA + 16);
    b1A.q[0] = *reinterpret_cast<const v4u*>(wcol1 + bA);
    b1A.q[1] = *reinterpret_cast<const v4u*>(wcol1 + bA + 16);
    // set B (k0+32)
    a0B.q[0] = *reinterpret_cast<const v4u*>(arow0 + bB);
    a0B.q[1] = *reinterpret_cast<const v4u*>(arow0 + bB + 16);
    a1B.q[0] = *reinterpret_cast<const v4u*>(arow1 + bB);
    a1B.q[1] = *reinterpret_cast<const v4u*>(arow1 + bB + 16);
    b0B.q[0] = *reinterpret_cast<const v4u*>(wcol0 + bB);
    b0B.q[1] = *reinterpret_cast<const v4u*>(wcol0 + bB + 16);
    b1B.q[0] = *reinterpret_cast<const v4u*>(wcol1 + bB);
    b1B.q[1] = *reinterpret_cast<const v4u*>(wcol1 + bB + 16);

    c00 = __builtin_amdgcn_wmma_f32_16x16x32_bf16(false, a0A.v, false, b0A.v, (short)0, c00, false, false);
    c01 = __builtin_amdgcn_wmma_f32_16x16x32_bf16(false, a0A.v, false, b1A.v, (short)0, c01, false, false);
    c10 = __builtin_amdgcn_wmma_f32_16x16x32_bf16(false, a1A.v, false, b0A.v, (short)0, c10, false, false);
    c11 = __builtin_amdgcn_wmma_f32_16x16x32_bf16(false, a1A.v, false, b1A.v, (short)0, c11, false, false);

    c00 = __builtin_amdgcn_wmma_f32_16x16x32_bf16(false, a0B.v, false, b0B.v, (short)0, c00, false, false);
    c01 = __builtin_amdgcn_wmma_f32_16x16x32_bf16(false, a0B.v, false, b1B.v, (short)0, c01, false, false);
    c10 = __builtin_amdgcn_wmma_f32_16x16x32_bf16(false, a1B.v, false, b0B.v, (short)0, c10, false, false);
    c11 = __builtin_amdgcn_wmma_f32_16x16x32_bf16(false, a1B.v, false, b1B.v, (short)0, c11, false, false);
  }

  // ---- epilogue: C/D layout -> elem r: M = r + 8*hi (within 16x16 tile)
  auto epil = [&](const v8f& c, int rbase, int col) {
    const float bv = bias ? bias[col] : 0.f;
    const float ew = extra_w ? extra_w[col] : 0.f;
    #pragma unroll
    for (int r = 0; r < 8; ++r) {
      int grow = rbase + r + hi * 8;
      if (grow < n) {
        float v = c[r] + bv;
        if (extra_s) v += extra_s[grow] * ew;
        if (act)     v = selu_f(v);
        if (res)     v += res[(long)grow * 128 + col];
        Y[(long)grow * 128 + col] = v;
      }
    }
  };
  epil(c00, row0 + r0,      col0);
  epil(c01, row0 + r0,      col0 + 16);
  epil(c10, row0 + r0 + 16, col0);
  epil(c11, row0 + r0 + 16, col0 + 16);
}

// W: K x 128 row-major f32  ->  Wt: 128 x K bf16 (column-major over W)
__global__ void prep_wt_kernel(const float* __restrict__ W, u16* __restrict__ Wt, int K) {
  int tid = blockIdx.x * 256 + threadIdx.x;
  if (tid >= K * 128) return;
  int n = tid / K;
  int k = tid - n * K;
  Wt[tid] = f2bf(W[(long)k * 128 + n]);
}

__global__ void zero_kernel(float* __restrict__ p, long n) {
  long i = (long)blockIdx.x * 256 + threadIdx.x;
  if (i < n) p[i] = 0.f;
}

__global__ void scatter_add_kernel(const float* __restrict__ e, const int* __restrict__ dst,
                                   int E, float* __restrict__ agg, float* __restrict__ cnt) {
  long i = (long)blockIdx.x * 256 + threadIdx.x;
  if (i >= (long)E * 128) return;
  int edge = (int)(i >> 7);
  int f    = (int)(i & 127);
  int d    = dst[edge];
  atomicAdd(&agg[(long)d * 128 + f], e[i]);
  if (f == 0) atomicAdd(&cnt[d], 1.f);
}

__global__ void mean_kernel(float* __restrict__ agg, const float* __restrict__ cnt, long n128) {
  long i = (long)blockIdx.x * 256 + threadIdx.x;
  if (i < n128) agg[i] /= fmaxf(cnt[i >> 7], 1.f);
}

// final 128 -> 4 projection (tiny FLOPs, plain VALU)
__global__ void out_proj_kernel(const float* __restrict__ v, const float* __restrict__ w,
                                const float* __restrict__ b, float* __restrict__ y, int n) {
  int i = blockIdx.x * 256 + threadIdx.x;
  if (i >= n * 4) return;
  int row = i >> 2;
  int c   = i & 3;
  float acc = b[c];
  const float* vr = v + (long)row * 128;
  #pragma unroll 8
  for (int k = 0; k < 128; ++k) acc += vr[k] * w[k * 4 + c];
  y[i] = acc;
}

// ---------------------------------------------------------------------------

extern "C" void kernel_launch(void* const* d_in, const int* in_sizes, int n_in,
                              void* d_out, int out_size, void* d_ws, size_t ws_size,
                              hipStream_t stream) {
  (void)in_sizes; (void)n_in; (void)out_size; (void)ws_size;

  const float* v_in     = (const float*)d_in[0];
  const float* c_coarse = (const float*)d_in[1];
  const float* c_fine   = (const float*)d_in[2];
  const float* e_coarse = (const float*)d_in[3];
  const float* e_fine   = (const float*)d_in[4];
  const float* rel_dist = (const float*)d_in[5];
  const int*   ei_c     = (const int*)d_in[6];
  const int*   ei_f     = (const int*)d_in[7];
  const int*   cluster  = (const int*)d_in[8];
  const float* in_node_w = (const float*)d_in[9];
  const float* in_node_b = (const float*)d_in[10];
  const float* cond_w    = (const float*)d_in[11];
  const float* cond_b    = (const float*)d_in[12];
  const float* edge_in_w = (const float*)d_in[13];
  const float* edge_in_b = (const float*)d_in[14];
  const float* bn_ew1 = (const float*)d_in[15];
  const float* bn_eb1 = (const float*)d_in[16];
  const float* bn_ew2 = (const float*)d_in[17];
  const float* bn_eb2 = (const float*)d_in[18];
  const float* bn_nw1 = (const float*)d_in[19];
  const float* bn_nb1 = (const float*)d_in[20];
  const float* bn_nw2 = (const float*)d_in[21];
  const float* bn_nb2 = (const float*)d_in[22];
  const float* unp_w1 = (const float*)d_in[23];
  const float* unp_b1 = (const float*)d_in[24];
  const float* unp_w2 = (const float*)d_in[25];
  const float* unp_b2 = (const float*)d_in[26];
  const float* up_ew1 = (const float*)d_in[27];
  const float* up_eb1 = (const float*)d_in[28];
  const float* up_ew2 = (const float*)d_in[29];
  const float* up_eb2 = (const float*)d_in[30];
  const float* up_nw1 = (const float*)d_in[31];
  const float* up_nb1 = (const float*)d_in[32];
  const float* up_nw2 = (const float*)d_in[33];
  const float* up_nb2 = (const float*)d_in[34];
  const float* out_w  = (const float*)d_in[35];
  const float* out_b  = (const float*)d_in[36];

  const int* src_c = ei_c;        const int* dst_c = ei_c + E_C;
  const int* src_f = ei_f;        const int* dst_f = ei_f + E_F;

  // ---- workspace carve-out
  char* ws = (char*)d_ws;
  size_t off = 0;
  auto take = [&](size_t bytes) -> char* {
    char* p = ws + off;
    off = (off + bytes + 255) & ~(size_t)255;
    return p;
  };
  auto takeWt = [&](int K) -> u16* { return (u16*)take((size_t)K * 128 * 2); };

  u16* wt_in     = takeWt(128);
  u16* wt_cond0  = takeWt(128);
  u16* wt_cond1  = takeWt(128);
  u16* wt_edge0  = takeWt(128);
  u16* wt_edge1  = takeWt(128);
  u16* wt_bn_ew1[2] = { takeWt(384), takeWt(384) };
  u16* wt_bn_ew2[2] = { takeWt(128), takeWt(128) };
  u16* wt_bn_nw1[2] = { takeWt(256), takeWt(256) };
  u16* wt_bn_nw2[2] = { takeWt(128), takeWt(128) };
  u16* wt_unp1   = takeWt(128);
  u16* wt_unp2   = takeWt(128);
  u16* wt_up_ew1[2] = { takeWt(384), takeWt(384) };
  u16* wt_up_ew2[2] = { takeWt(128), takeWt(128) };
  u16* wt_up_nw1[2] = { takeWt(256), takeWt(256) };
  u16* wt_up_nw2[2] = { takeWt(128), takeWt(128) };

  float* v_c = (float*)take((size_t)N_C * 128 * 4);
  float* e_c = (float*)take((size_t)E_C * 128 * 4);
  float* v_f = (float*)take((size_t)N_F * 128 * 4);
  float* e_f = (float*)take((size_t)E_F * 128 * 4);
  float* H   = (float*)take((size_t)E_F * 128 * 4);   // hidden (max size)
  float* agg = (float*)take((size_t)N_F * 128 * 4);   // also reused as c_fine proj
  float* cnt = (float*)take((size_t)N_F * 4);
  float* cf  = agg;  // lifetime-disjoint alias (consumed before first fine scatter)

  // ---- helpers
  auto prep = [&](const float* Wsrc, u16* Wdst, int K) {
    int tot = K * 128;
    prep_wt_kernel<<<dim3((tot + 255) / 256), dim3(256), 0, stream>>>(Wsrc, Wdst, K);
  };
  auto gemm = [&](const float* X0, const int* i0, const float* X1, const int* i1,
                  const float* X2, const int* i2, int n, int K,
                  const u16* Wt, const float* bias,
                  const float* es, const float* ew, const float* res,
                  float* Y, int act) {
    gemm128_wmma<<<dim3((n + 63) / 64), dim3(256), 0, stream>>>(
        X0, i0, X1, i1, X2, i2, n, K, Wt, bias, es, ew, res, Y, act);
  };
  auto zero = [&](float* p, long n) {
    zero_kernel<<<dim3((unsigned)((n + 255) / 256)), dim3(256), 0, stream>>>(p, n);
  };
  auto scatter_mean = [&](const float* e, const int* dst, int E, int N) {
    zero(agg, (long)N * 128);
    zero(cnt, N);
    long tot = (long)E * 128;
    scatter_add_kernel<<<dim3((unsigned)((tot + 255) / 256)), dim3(256), 0, stream>>>(
        e, dst, E, agg, cnt);
    mean_kernel<<<dim3((unsigned)(((long)N * 128 + 255) / 256)), dim3(256), 0, stream>>>(
        agg, cnt, (long)N * 128);
  };

  // ---- weight prep (bf16 + transpose); all tiny, L2 resident afterwards
  prep(in_node_w, wt_in, 128);
  prep(cond_w,                wt_cond0, 128);
  prep(cond_w + 128 * 128,    wt_cond1, 128);
  prep(edge_in_w,             wt_edge0, 128);
  prep(edge_in_w + 128 * 128, wt_edge1, 128);
  for (int i = 0; i < 2; ++i) {
    prep(bn_ew1 + (size_t)i * 384 * 128, wt_bn_ew1[i], 384);
    prep(bn_ew2 + (size_t)i * 128 * 128, wt_bn_ew2[i], 128);
    prep(bn_nw1 + (size_t)i * 256 * 128, wt_bn_nw1[i], 256);
    prep(bn_nw2 + (size_t)i * 128 * 128, wt_bn_nw2[i], 128);
    prep(up_ew1 + (size_t)i * 384 * 128, wt_up_ew1[i], 384);
    prep(up_ew2 + (size_t)i * 128 * 128, wt_up_ew2[i], 128);
    prep(up_nw1 + (size_t)i * 256 * 128, wt_up_nw1[i], 256);
    prep(up_nw2 + (size_t)i * 128 * 128, wt_up_nw2[i], 128);
  }
  prep(unp_w1, wt_unp1, 128);   // first 128 rows; row 128 handled as rank-1 epilogue
  prep(unp_w2, wt_unp2, 128);

  // ---- input projection + coarse conditioning
  gemm(v_in, 0, 0, 0, 0, 0, N_C, 128, wt_in, in_node_b, 0, 0, 0,   v_c, 0);
  gemm(c_coarse, 0, 0, 0, 0, 0, N_C, 128, wt_cond0, cond_b, 0, 0, v_c, v_c, 0);
  gemm(e_coarse, 0, 0, 0, 0, 0, E_C, 128, wt_edge0, edge_in_b, 0, 0, 0, e_c, 0);

  // ---- bottleneck interaction blocks (coarse graph)
  for (int i = 0; i < 2; ++i) {
    gemm(e_c, 0, v_c, src_c, v_c, dst_c, E_C, 384,
         wt_bn_ew1[i], bn_eb1 + i * 128, 0, 0, 0, H, 1);
    gemm(H, 0, 0, 0, 0, 0, E_C, 128,
         wt_bn_ew2[i], bn_eb2 + i * 128, 0, 0, e_c, e_c, 0);
    scatter_mean(e_c, dst_c, E_C, N_C);
    gemm(v_c, 0, agg, 0, 0, 0, N_C, 256,
         wt_bn_nw1[i], bn_nb1 + i * 128, 0, 0, 0, H, 1);
    gemm(H, 0, 0, 0, 0, 0, N_C, 128,
         wt_bn_nw2[i], bn_nb2 + i * 128, 0, 0, v_c, v_c, 0);
  }

  // ---- up block: skip conditioning, edge projection, unpool via cluster gather
  gemm(c_fine, 0, 0, 0, 0, 0, N_F, 128, wt_cond1, cond_b + 128, 0, 0, 0, cf, 0);
  gemm(e_fine, 0, 0, 0, 0, 0, E_F, 128, wt_edge1, edge_in_b + 128, 0, 0, 0, e_f, 0);
  gemm(v_c, cluster, 0, 0, 0, 0, N_F, 128,
       wt_unp1, unp_b1, rel_dist, unp_w1 + 128 * 128, 0, H, 1);
  gemm(H, 0, 0, 0, 0, 0, N_F, 128, wt_unp2, unp_b2, 0, 0, cf, v_f, 0);

  // ---- up interaction blocks (fine graph)
  for (int i = 0; i < 2; ++i) {
    gemm(e_f, 0, v_f, src_f, v_f, dst_f, E_F, 384,
         wt_up_ew1[i], up_eb1 + i * 128, 0, 0, 0, H, 1);
    gemm(H, 0, 0, 0, 0, 0, E_F, 128,
         wt_up_ew2[i], up_eb2 + i * 128, 0, 0, e_f, e_f, 0);
    scatter_mean(e_f, dst_f, E_F, N_F);
    gemm(v_f, 0, agg, 0, 0, 0, N_F, 256,
         wt_up_nw1[i], up_nb1 + i * 128, 0, 0, 0, H, 1);
    gemm(H, 0, 0, 0, 0, 0, N_F, 128,
         wt_up_nw2[i], up_nb2 + i * 128, 0, 0, v_f, v_f, 0);
  }

  // ---- output projection 128 -> 4
  out_proj_kernel<<<dim3((N_F * 4 + 255) / 256), dim3(256), 0, stream>>>(
      v_f, out_w, out_b, (float*)d_out, N_F);
}
